// MMs4d_9698036155139
// MI455X (gfx1250) — compile-verified, matrix operand order checked by reading, not run on hardware
//
#include <hip/hip_runtime.h>
#include <hip/hip_bf16.h>

typedef __attribute__((ext_vector_type(16))) _Float16 v16h;
typedef __attribute__((ext_vector_type(8)))  float    v8f;

#define M_TOK 32768   // B*NODE*L = 128*256
#define SEQ_L 256
#define NSEQ  128

__device__ __forceinline__ float silu_f(float v) {
    return v / (1.f + __expf(-v));
}
__device__ __forceinline__ float softplus_f(float v) {
    return (v > 20.f) ? v : __logf(1.f + __expf(v));
}

// --- CDNA5 async global->LDS copy (ASYNCcnt-tracked) ------------------------
__device__ __forceinline__ void async_ld32(uint32_t lds_off, const float* g) {
    asm volatile("global_load_async_to_lds_b32 %0, %1, off"
                 :: "v"(lds_off), "v"(g) : "memory");
}
__device__ __forceinline__ void wait_async0() {
    asm volatile("s_wait_asynccnt 0x0" ::: "memory");
}
__device__ __forceinline__ uint32_t lds_off32(const void* p) {
    // LDS aperture lives in ADDR[63:32]; low 32 bits == wave-relative LDS addr
    return (uint32_t)(uintptr_t)p;
}

// ---------------------------------------------------------------------------
// Generic WMMA GEMM (K = 64 compile-time): C = act( A[Mx64] @ W[Nx64]^T + b )
// One wave computes one 16x16 tile. grid.x = M/16, grid.y = N/64 (4 waves/blk)
// ACT: 0 = none, 1 = silu, 2 = softplus
// ---------------------------------------------------------------------------
template <int ACT>
__global__ __launch_bounds__(128) void gemm_wmma(
    const float* __restrict__ A, const float* __restrict__ W,
    const float* __restrict__ bias, float* __restrict__ C, int N)
{
    constexpr int K = 64;
    const int wave = threadIdx.x >> 5;
    const int lane = threadIdx.x & 31;
    const int mtile = blockIdx.x * 16;
    const int ntile = (blockIdx.y * 4 + wave) * 16;
    const int mrow  = mtile + (lane & 15);
    const int nrow  = ntile + (lane & 15);
    const int kg    = (lane >> 4) * 8;   // 0 or 8 : lane k-group

    v8f acc = {};
#pragma unroll
    for (int k0 = 0; k0 < K; k0 += 32) {
        const float* Ar = A + (size_t)mrow * K + k0 + kg;
        const float* Wr = W + (size_t)nrow * K + k0 + kg;
        v16h a, b;
#pragma unroll
        for (int i = 0; i < 8; ++i) {
            a[i]     = (_Float16)Ar[i];        // k = k0+kg+i
            a[8 + i] = (_Float16)Ar[16 + i];   // k = k0+16+kg+i
            b[i]     = (_Float16)Wr[i];
            b[8 + i] = (_Float16)Wr[16 + i];
        }
        acc = __builtin_amdgcn_wmma_f32_16x16x32_f16(
            false, a, false, b, (short)0, acc, false, false);
    }

    const int mbase = mtile + 8 * (lane >> 4);
    const int nn    = ntile + (lane & 15);
    const float bv  = bias ? bias[nn] : 0.f;
#pragma unroll
    for (int j = 0; j < 8; ++j) {
        float v = acc[j] + bv;
        if (ACT == 1) v = silu_f(v);
        if (ACT == 2) v = softplus_f(v);
        C[(size_t)(mbase + j) * N + nn] = v;
    }
}

// ---------------------------------------------------------------------------
// Input projection with fused gather-transpose:
//   token m = r*256 + t, r = b*32 + node
//   A[m][k] = inputs[((b*128 + k)*32 + node)*256 + t]
//   C = A @ w_in^T + b_in    (N = 64, K = 128)
// ---------------------------------------------------------------------------
__global__ __launch_bounds__(128) void gemm_in_wmma(
    const float* __restrict__ In, const float* __restrict__ W,
    const float* __restrict__ bias, float* __restrict__ C)
{
    const int K = 128, N = 64;
    const int wave = threadIdx.x >> 5;
    const int lane = threadIdx.x & 31;
    const int mtile = blockIdx.x * 16;
    const int ntile = wave * 16;
    const int mrow  = mtile + (lane & 15);
    const int nrow  = ntile + (lane & 15);
    const int kg    = (lane >> 4) * 8;

    const int r = mrow >> 8, t = mrow & 255;
    const int b = r >> 5, node = r & 31;
    const float* ibase = In + (size_t)b * 128 * 8192 + node * 256 + t;

    v8f acc = {};
#pragma unroll
    for (int k0 = 0; k0 < K; k0 += 32) {
        const float* Wr = W + (size_t)nrow * K + k0 + kg;
        v16h a, bb;
#pragma unroll
        for (int i = 0; i < 8; ++i) {
            a[i]      = (_Float16)ibase[(size_t)(k0 + kg + i) * 8192];
            a[8 + i]  = (_Float16)ibase[(size_t)(k0 + kg + 16 + i) * 8192];
            bb[i]     = (_Float16)Wr[i];
            bb[8 + i] = (_Float16)Wr[16 + i];
        }
        acc = __builtin_amdgcn_wmma_f32_16x16x32_f16(
            false, a, false, bb, (short)0, acc, false, false);
    }

    const int mbase = mtile + 8 * (lane >> 4);
    const int nn    = ntile + (lane & 15);
    const float bv  = bias[nn];
#pragma unroll
    for (int j = 0; j < 8; ++j)
        C[(size_t)(mbase + j) * N + nn] = acc[j] + bv;
}

// ---------------------------------------------------------------------------
// Fold dt_proj_w (64x4) @ x_proj_w[:4] (4x64) -> Wd (64x64)
// ---------------------------------------------------------------------------
__global__ void prep_wd(const float* __restrict__ dtw,
                        const float* __restrict__ xpw,
                        float* __restrict__ Wd)
{
    int idx = blockIdx.x * 256 + threadIdx.x;
    if (idx >= 64 * 64) return;
    int d = idx >> 6, k = idx & 63;
    float s = 0.f;
#pragma unroll
    for (int rr = 0; rr < 4; ++rr)
        s += dtw[d * 4 + rr] * xpw[rr * 64 + k];
    Wd[idx] = s;
}

// ---------------------------------------------------------------------------
// Causal depthwise conv (d_conv=2) + SiLU.  xm = xz[:, :64]
// ---------------------------------------------------------------------------
__global__ void conv_silu_kernel(const float* __restrict__ xz,
                                 const float* __restrict__ cw,
                                 const float* __restrict__ cb,
                                 float* __restrict__ xc)
{
    int idx = blockIdx.x * 256 + threadIdx.x;      // over M_TOK*64
    if (idx >= M_TOK * 64) return;
    int m = idx >> 6, d = idx & 63;
    int t = m & 255;
    float cur  = xz[(size_t)m * 128 + d];
    float prev = (t > 0) ? xz[(size_t)(m - 1) * 128 + d] : 0.f;
    float v = prev * cw[d * 2 + 0] + cur * cw[d * 2 + 1] + cb[d];
    xc[idx] = silu_f(v);
}

// ---------------------------------------------------------------------------
// Selective scan: one workgroup per sequence (256 thr = 8 wave32).
// thread = d*4 + sg; owns h[d][sg*32 .. sg*32+31] in registers.
// Per-step operands (B:128 | C:128 | delta:64 | x:64 floats) are double-
// buffered in LDS via async global->LDS copies (ASYNCcnt), so the next
// step's fetch overlaps the current step's recurrence math.
// Epilogue fused: ymix = (y + xc*D) * silu(z)
// ---------------------------------------------------------------------------
__global__ __launch_bounds__(256) void scan_kernel(
    const float* __restrict__ delta, const float* __restrict__ BC,
    const float* __restrict__ xc,    const float* __restrict__ xz,
    const float* __restrict__ A_log, const float* __restrict__ Dvec,
    float* __restrict__ ymix)
{
    // [0,128)=B  [128,256)=C  [256,320)=delta  [320,384)=x
    __shared__ float sBuf[2][384];
    const int r   = blockIdx.x;
    const int tid = threadIdx.x;
    const int d   = tid >> 2;
    const int sg  = tid & 3;
    const int sb  = sg * 32;

    float Areg[32], h[32];
#pragma unroll
    for (int i = 0; i < 32; ++i) {
        Areg[i] = -__expf(A_log[d * 128 + sb + i]);
        h[i] = 0.f;
    }
    const float Dd = Dvec[d];
    const size_t mbase = (size_t)r * SEQ_L;

    // issue async loads for one timestep into buffer `buf`
    auto issue = [&](int t, int buf) {
        const size_t m = mbase + t;
        if (tid < 128) {
            async_ld32(lds_off32(&sBuf[buf][tid]),       BC + m * 256 + tid);
            async_ld32(lds_off32(&sBuf[buf][128 + tid]), BC + m * 256 + 128 + tid);
        } else if (tid < 192) {
            const int q = tid - 128;
            async_ld32(lds_off32(&sBuf[buf][256 + q]), delta + m * 64 + q);
        } else {
            const int q = tid - 192;
            async_ld32(lds_off32(&sBuf[buf][320 + q]), xc + m * 64 + q);
        }
    };

    int cur = 0;
    issue(0, 0);

    for (int t = 0; t < SEQ_L; ++t) {
        wait_async0();        // this wave's async copies into buf[cur] landed
        __syncthreads();      // all waves' copies visible; prev-step reads done

        if (t + 1 < SEQ_L) issue(t + 1, cur ^ 1);   // overlap with compute

        const float* S = sBuf[cur];
        const float dlt = S[256 + d];
        const float xt  = S[320 + d];
        const float dbx = dlt * xt;
        float acc = 0.f;
#pragma unroll
        for (int i = 0; i < 32; ++i) {
            float dA = __expf(dlt * Areg[i]);
            h[i] = dA * h[i] + dbx * S[sb + i];
            acc += h[i] * S[128 + sb + i];
        }
        // reduce over the 4 lanes sharing d (adjacent lanes in a wave32)
        acc += __shfl_xor(acc, 1, 32);
        acc += __shfl_xor(acc, 2, 32);
        if (sg == 0) {
            const size_t m = mbase + t;
            float zv = xz[m * 128 + 64 + d];
            ymix[m * 64 + d] = (acc + xt * Dd) * silu_f(zv);
        }
        cur ^= 1;
    }
}

// ---------------------------------------------------------------------------
// Per-sequence LayerNorm over (L, N_OUT) + scale/shift + transposed scatter
// out[((b*128+o)*32+node)*256 + t]
// ---------------------------------------------------------------------------
__global__ __launch_bounds__(256) void ln_kernel(
    const float* __restrict__ x3, const float* __restrict__ lnw,
    const float* __restrict__ lnb, float* __restrict__ out)
{
    const int r = blockIdx.x, tid = threadIdx.x;
    const float* xr = x3 + (size_t)r * SEQ_L * 128;

    float s = 0.f, s2 = 0.f;
    for (int e = tid; e < SEQ_L * 128; e += 256) {
        float v = xr[e];
        s += v; s2 += v * v;
    }
    __shared__ float rs[256], rs2[256];
    rs[tid] = s; rs2[tid] = s2;
    __syncthreads();
    for (int off = 128; off > 0; off >>= 1) {
        if (tid < off) { rs[tid] += rs[tid + off]; rs2[tid] += rs2[tid + off]; }
        __syncthreads();
    }
    const float inv_n = 1.f / (SEQ_L * 128.f);
    const float mu   = rs[0] * inv_n;
    const float var  = rs2[0] * inv_n - mu * mu;
    const float rsig = rsqrtf(var + 1e-5f);

    const int b = r >> 5, node = r & 31;
    float* ob = out + (size_t)b * 1048576 + node * 256;
    for (int e = tid; e < SEQ_L * 128; e += 256) {
        int t = e >> 7, o = e & 127;
        float v = (xr[e] - mu) * rsig * lnw[t * 128 + o] + lnb[t * 128 + o];
        ob[(size_t)o * 8192 + t] = v;
    }
}

// ---------------------------------------------------------------------------
extern "C" void kernel_launch(void* const* d_in, const int* in_sizes, int n_in,
                              void* d_out, int out_size, void* d_ws, size_t ws_size,
                              hipStream_t stream)
{
    (void)in_sizes; (void)n_in; (void)out_size; (void)ws_size;
    const float* inp   = (const float*)d_in[0];
    const float* w_in  = (const float*)d_in[1];
    const float* b_in  = (const float*)d_in[2];
    const float* inpw  = (const float*)d_in[3];
    const float* convw = (const float*)d_in[4];
    const float* convb = (const float*)d_in[5];
    const float* xpw   = (const float*)d_in[6];
    const float* dtw   = (const float*)d_in[7];
    const float* dtb   = (const float*)d_in[8];
    const float* alog  = (const float*)d_in[9];
    const float* Dvec  = (const float*)d_in[10];
    const float* outpw = (const float*)d_in[11];
    const float* wout  = (const float*)d_in[12];
    const float* bout  = (const float*)d_in[13];
    const float* lnw   = (const float*)d_in[14];
    const float* lnb   = (const float*)d_in[15];
    float* out = (float*)d_out;

    const size_t M = M_TOK;
    float* ws   = (float*)d_ws;
    float* x_   = ws;                 // M*64   x after input proj
    float* xz   = x_   + M * 64;      // M*128  xm | z
    float* xc   = xz   + M * 128;     // M*64   conv+silu
    float* dlt  = xc   + M * 64;      // M*64   softplus delta
    float* BCb  = dlt  + M * 64;      // M*256  B | C
    float* ymix = BCb  + M * 256;     // M*64   scan output (gated)
    float* Wd   = ymix + M * 64;      // 64*64  folded dt projection
    float* mam  = x_;                 // alias: x_ dead after in_proj GEMM
    float* x3   = BCb;                // alias: BC dead after scan

    const int MT = M_TOK / 16;        // 2048 M-tiles

    prep_wd<<<16, 256, 0, stream>>>(dtw, xpw, Wd);

    // x = gather(inputs) @ w_in^T + b_in                    [N=64, K=128]
    gemm_in_wmma<<<dim3(MT, 1), 128, 0, stream>>>(inp, w_in, b_in, x_);

    // xz = x @ in_proj_w^T                                   [N=128, K=64]
    gemm_wmma<0><<<dim3(MT, 2), 128, 0, stream>>>(x_, inpw, nullptr, xz, 128);

    // xc = silu(conv(xm) + conv_b)
    conv_silu_kernel<<<(M_TOK * 64) / 256, 256, 0, stream>>>(xz, convw, convb, xc);

    // delta = softplus(xc @ Wd^T + dt_proj_b)                [N=64, K=64]
    gemm_wmma<2><<<dim3(MT, 1), 128, 0, stream>>>(xc, Wd, dtb, dlt, 64);

    // [B|C] = xc @ x_proj_w[4:]^T                            [N=256, K=64]
    gemm_wmma<0><<<dim3(MT, 4), 128, 0, stream>>>(xc, xpw + 4 * 64, nullptr, BCb, 256);

    // selective scan + gating, one workgroup per sequence (async double-buffer)
    scan_kernel<<<NSEQ, 256, 0, stream>>>(dlt, BCb, xc, xz, alog, Dvec, ymix);

    // mam = silu(ymix @ out_proj_w^T)                        [N=64, K=64]
    gemm_wmma<1><<<dim3(MT, 1), 128, 0, stream>>>(ymix, outpw, nullptr, mam, 64);

    // x3 = mam @ w_out^T + b_out                             [N=128, K=64]
    gemm_wmma<0><<<dim3(MT, 2), 128, 0, stream>>>(mam, wout, bout, x3, 128);

    // per-sequence layernorm + output transpose
    ln_kernel<<<NSEQ, 256, 0, stream>>>(x3, lnw, lnb, out);
}
